// CrossAttention_65257733095826
// MI455X (gfx1250) — matrix operand order repaired and checked by reading, not compiled, not run on hardware
//
#include <hip/hip_runtime.h>
#include <hip/hip_bf16.h>

typedef __attribute__((ext_vector_type(16))) _Float16 v16h;
typedef __attribute__((ext_vector_type(8)))  _Float16 v8h;
typedef __attribute__((ext_vector_type(8)))  float    v8f;

#define DEVINL __device__ __forceinline__

constexpr int BN_ = 4;            // batch
constexpr int C_  = 256;          // channels
constexpr int HH  = 48, WW = 48;
constexpr int P_  = HH * WW;      // 2304 pixels
constexpr int HEADS_ = 4, D_ = 64;
constexpr float EPS_ = 1e-5f;

// ---------------------------------------------------------------- WMMA helpers

DEVINL v8f wmma16(v16h a, v16h b, v8f c) {
  // D = A(16x32 f16) * B(32x16 f16) + C(16x16 f32)
  return __builtin_amdgcn_wmma_f32_16x16x32_f16(
      /*neg_a=*/false, a, /*neg_b=*/false, b,
      /*c_mod=*/(short)0, c, /*reuse_a=*/false, /*reuse_b=*/false);
}

DEVINL v16h cat8(v8h lo, v8h hi) {
  return __builtin_shufflevector(lo, hi, 0,1,2,3,4,5,6,7,8,9,10,11,12,13,14,15);
}

// A fragment: 16x32 f16. Lane L holds row m=L%16; VGPR0-3 = K in [(L/16)*8,+8),
// VGPR4-7 = K in [16+(L/16)*8,+8). base points at (row0, K0), ld in halves.
DEVINL v16h load_a(const _Float16* base, int ld) {
  int lane = threadIdx.x & 31;
  int m = lane & 15, kb = (lane >> 4) << 3;
  v8h lo = *(const v8h*)(base + (size_t)m * ld + kb);
  v8h hi = *(const v8h*)(base + (size_t)m * ld + kb + 16);
  return cat8(lo, hi);
}

// B fragment: 32x16 f16. Lane L holds column n=L%16; VGPR0-3 = K in [(L/16)*8,+8),
// VGPR4-7 = K in [16+(L/16)*8,+8). Source layout: row index == B column, K contiguous.
DEVINL v16h load_b(const _Float16* base, int ld) {
  int lane = threadIdx.x & 31;
  int n = lane & 15, kb = (lane >> 4) << 3;
  v8h lo = *(const v8h*)(base + (size_t)n * ld + kb);
  v8h hi = *(const v8h*)(base + (size_t)n * ld + kb + 16);
  return cat8(lo, hi);
}

// ---------------------------------------------------------------- pack kernels

// NCHW f32 -> [b][p][c] f16 (B-side layout for the 1x1 convs)
__global__ void pack_xT_kernel(const float* __restrict__ x, _Float16* __restrict__ xT) {
  int total = BN_ * P_ * C_;
  for (int i = blockIdx.x * blockDim.x + threadIdx.x; i < total;
       i += gridDim.x * blockDim.x) {
    int c = i & (C_ - 1);
    int p = (i >> 8) % P_;
    int b = i / (P_ * C_);
    xT[i] = (_Float16)x[((size_t)b * C_ + c) * P_ + p];
  }
}

__global__ void cvt_w16_kernel(const float* __restrict__ w, _Float16* __restrict__ o, int n) {
  for (int i = blockIdx.x * blockDim.x + threadIdx.x; i < n;
       i += gridDim.x * blockDim.x)
    o[i] = (_Float16)w[i];
}

// ---------------------------------------------------------- 1x1 conv + BN GEMM

// out[o,p] = sum_c W[o,c]*x[c,p]; BN folded into epilogue. One wave = one
// 16(M=out-ch) x 16(N=pixel) tile; K loop over 256 input channels (8 WMMAs).
__global__ void __launch_bounds__(256)
conv1x1_bn_wmma(const _Float16* __restrict__ xT,   // [B][P][C]
                const _Float16* __restrict__ w16,  // [C][C] row-major (o,c)
                const float* __restrict__ gg, const float* __restrict__ bb,
                const float* __restrict__ mm, const float* __restrict__ vv,
                _Float16* __restrict__ out_nd,     // [B][HEADS][P][D] f16 or null
                _Float16* __restrict__ out_t16,    // [B][C][P]  f16 (NCHW) or null
                float*    __restrict__ out_f32)    // [B][C][P]  f32 (NCHW) or null
{
  int wave = (blockIdx.x * blockDim.x + threadIdx.x) >> 5;
  int lane = threadIdx.x & 31;
  constexpr int NT = P_ / 16;   // 144 pixel tiles
  constexpr int MT = C_ / 16;   // 16 out-channel tiles
  int nt = wave % NT;
  int rest = wave / NT;
  int mt = rest % MT;
  int b  = rest / MT;
  if (b >= BN_) return;

  const _Float16* xb = xT  + (size_t)b * P_ * C_ + (size_t)(nt * 16) * C_;
  const _Float16* wb = w16 + (size_t)(mt * 16) * C_;

  v8f acc = {};
#pragma unroll
  for (int c0 = 0; c0 < C_; c0 += 32) {
    v16h af = load_a(wb + c0, C_);
    v16h bf = load_b(xb + c0, C_);
    acc = wmma16(af, bf, acc);
  }

  int half_ = lane >> 4, col = lane & 15;
  int p = nt * 16 + col;
#pragma unroll
  for (int r = 0; r < 8; ++r) {
    int o = mt * 16 + r + 8 * half_;
    float s = gg[o] * rsqrtf(vv[o] + EPS_);
    float t = bb[o] - mm[o] * s;
    float val = acc[r] * s + t;
    size_t inchw = ((size_t)b * C_ + o) * P_ + p;
    if (out_f32) out_f32[inchw] = val;
    if (out_t16) out_t16[inchw] = (_Float16)val;
    if (out_nd) {
      int hd = o >> 6, d = o & 63;
      out_nd[(((size_t)b * HEADS_ + hd) * P_ + p) * (size_t)D_ + d] = (_Float16)val;
    }
  }
}

// ---------------------------------------------------------- flash attention

// One wave per (b, head, 16-query block). Key blocks of 32; online softmax.
// qh/kh: [B][H][N][64] f16 ; vT: [B][H][64][N] f16 ; yout: [B][P][C] f32.
__global__ void __launch_bounds__(128)
flash_attn_wmma(const _Float16* __restrict__ qh, const _Float16* __restrict__ kh,
                const _Float16* __restrict__ vT, float* __restrict__ yout)
{
  __shared__ __align__(32) _Float16 lds[4][16 * 32];
  int wslot = threadIdx.x >> 5;
  int lane  = threadIdx.x & 31;
  int wave  = blockIdx.x * (blockDim.x >> 5) + wslot;
  constexpr int QB = P_ / 16;   // 144 query blocks
  if (wave >= BN_ * HEADS_ * QB) return;
  int qb = wave % QB;
  int rest = wave / QB;
  int h = rest % HEADS_;
  int b = rest / HEADS_;

  const _Float16* qbase = qh + ((size_t)b * HEADS_ + h) * P_ * D_;
  const _Float16* kbase = kh + ((size_t)b * HEADS_ + h) * P_ * D_;
  const _Float16* vbase = vT + ((size_t)b * HEADS_ + h) * D_ * P_;

  // Q fragments for this query block (held across whole K loop)
  v16h a0 = load_a(qbase + (size_t)(qb * 16) * D_ + 0,  D_);
  v16h a1 = load_a(qbase + (size_t)(qb * 16) * D_ + 32, D_);

  float Mrow[8], lrow[8];
  v8f O[4];
  v8f zf = {};
#pragma unroll
  for (int r = 0; r < 8; ++r) { Mrow[r] = -1e30f; lrow[r] = 0.f; }
#pragma unroll
  for (int t = 0; t < 4; ++t) O[t] = zf;

  int half_ = lane >> 4, col = lane & 15;
  _Float16* L = &lds[wslot][0];
  const float scale = 0.125f;  // 64^-0.5

  for (int kb = 0; kb < P_ / 32; ++kb) {
    int n0 = kb * 32;
    // S = Q * K^T for 32 keys -> two 16x16 f32 tiles
    v16h b00 = load_b(kbase + (size_t)n0 * D_ + 0,  D_);
    v16h b01 = load_b(kbase + (size_t)n0 * D_ + 32, D_);
    v16h b10 = load_b(kbase + (size_t)(n0 + 16) * D_ + 0,  D_);
    v16h b11 = load_b(kbase + (size_t)(n0 + 16) * D_ + 32, D_);
    v8f s0 = wmma16(a0, b00, zf); s0 = wmma16(a1, b01, s0);
    v8f s1 = wmma16(a0, b10, zf); s1 = wmma16(a1, b11, s1);

    // online softmax: per-row (C layout: VGPR r, lanes<16 -> row r, else r+8)
#pragma unroll
    for (int r = 0; r < 8; ++r) {
      float x0 = s0[r] * scale, x1 = s1[r] * scale;
      float t = fmaxf(x0, x1);
#pragma unroll
      for (int msk = 1; msk < 16; msk <<= 1) t = fmaxf(t, __shfl_xor(t, msk, 32));
      float Mn = fmaxf(Mrow[r], t);
      float alpha = __expf(Mrow[r] - Mn);
      Mrow[r] = Mn;
      float p0 = __expf(x0 - Mn), p1 = __expf(x1 - Mn);
      float rs = p0 + p1;
#pragma unroll
      for (int msk = 1; msk < 16; msk <<= 1) rs += __shfl_xor(rs, msk, 32);
      lrow[r] = lrow[r] * alpha + rs;
#pragma unroll
      for (int t4 = 0; t4 < 4; ++t4) O[t4][r] *= alpha;
      // stage P (f16) row-major 16x32 into this wave's LDS slice
      int mrow = r + 8 * half_;
      L[mrow * 32 + col]      = (_Float16)p0;
      L[mrow * 32 + 16 + col] = (_Float16)p1;
    }
    // same-wave LDS RAW: wait for DS stores before transposed A-fragment reads
    asm volatile("s_wait_dscnt 0" ::: "memory");

    // A fragment of P from LDS (row m = lane%16, K chunks per A layout)
    v8h lo = *(const v8h*)(L + col * 32 + (half_ << 3));
    v8h hi = *(const v8h*)(L + col * 32 + 16 + (half_ << 3));
    v16h ap = cat8(lo, hi);

    // O += P * V  (B from vT: source row = head-dim, positions contiguous)
#pragma unroll
    for (int t4 = 0; t4 < 4; ++t4) {
      v16h bv = load_b(vbase + (size_t)(t4 * 16) * P_ + n0, P_);
      O[t4] = wmma16(ap, bv, O[t4]);
    }
  }

  // epilogue: normalize and store y merged as [b][p][c] f32
#pragma unroll
  for (int r = 0; r < 8; ++r) {
    float inv = 1.0f / lrow[r];
    int m = qb * 16 + r + 8 * half_;
#pragma unroll
    for (int t4 = 0; t4 < 4; ++t4)
      yout[((size_t)b * P_ + m) * C_ + h * 64 + t4 * 16 + col] = O[t4][r] * inv;
  }
}

// ------------------------------------------- depthwise 3x3 + BN + add + pack

__global__ void pe_dwconv_add_pack(const float* __restrict__ qf,  // [B][C][P] f32
                                   const float* __restrict__ pw,  // [C][3][3]
                                   const float* __restrict__ gg, const float* __restrict__ bb,
                                   const float* __restrict__ mm, const float* __restrict__ vv,
                                   const float* __restrict__ yattn, // [B][P][C] f32
                                   _Float16* __restrict__ xT2)      // [B][P][C] f16
{
  int total = BN_ * C_ * P_;
  for (int i = blockIdx.x * blockDim.x + threadIdx.x; i < total;
       i += gridDim.x * blockDim.x) {
    int p = i % P_;
    int rest = i / P_;
    int c = rest % C_;
    int b = rest / C_;
    int ph = p / WW, pwid = p % WW;
    const float* src = qf + ((size_t)b * C_ + c) * P_;
    float acc = 0.f;
#pragma unroll
    for (int di = 0; di < 3; ++di) {
      int hh = ph + di - 1;
      if (hh < 0 || hh >= HH) continue;
#pragma unroll
      for (int dj = 0; dj < 3; ++dj) {
        int ww2 = pwid + dj - 1;
        if (ww2 < 0 || ww2 >= WW) continue;
        acc += src[hh * WW + ww2] * pw[c * 9 + di * 3 + dj];
      }
    }
    float s = gg[c] * rsqrtf(vv[c] + EPS_);
    float t = bb[c] - mm[c] * s;
    float val = acc * s + t + yattn[((size_t)b * P_ + p) * C_ + c];
    xT2[((size_t)b * P_ + p) * C_ + c] = (_Float16)val;
  }
}

// ---------------------------------------------------------------- launcher

extern "C" void kernel_launch(void* const* d_in, const int* in_sizes, int n_in,
                              void* d_out, int out_size, void* d_ws, size_t ws_size,
                              hipStream_t stream) {
  (void)in_sizes; (void)n_in; (void)out_size; (void)ws_size;
  const float* q   = (const float*)d_in[0];
  const float* k   = (const float*)d_in[1];
  const float* v   = (const float*)d_in[2];
  const float* wq  = (const float*)d_in[3];
  const float* wk  = (const float*)d_in[4];
  const float* wv  = (const float*)d_in[5];
  const float* wpj = (const float*)d_in[6];
  const float* pew = (const float*)d_in[7];
  const float* qg = (const float*)d_in[8],  *qb2 = (const float*)d_in[9],
             * qm = (const float*)d_in[10], *qv2 = (const float*)d_in[11];
  const float* kg = (const float*)d_in[12], *kb2 = (const float*)d_in[13],
             * km = (const float*)d_in[14], *kv2 = (const float*)d_in[15];
  const float* vg = (const float*)d_in[16], *vb2 = (const float*)d_in[17],
             * vm = (const float*)d_in[18], *vv2 = (const float*)d_in[19];
  const float* pg = (const float*)d_in[20], *pb2 = (const float*)d_in[21],
             * pm = (const float*)d_in[22], *pv2 = (const float*)d_in[23];
  const float* jg = (const float*)d_in[24], *jb2 = (const float*)d_in[25],
             * jm = (const float*)d_in[26], *jv2 = (const float*)d_in[27];

  const size_t NE = (size_t)BN_ * C_ * P_;  // 2,359,296 elements
  char* ws = (char*)d_ws;
  size_t off = 0;
  auto alloc = [&](size_t bytes) -> void* {
    void* p = ws + off;
    off = (off + bytes + 255) & ~(size_t)255;
    return p;
  };
  _Float16* xTq  = (_Float16*)alloc(NE * 2);
  _Float16* xTk  = (_Float16*)alloc(NE * 2);
  _Float16* xTv  = (_Float16*)alloc(NE * 2);
  _Float16* qhB  = (_Float16*)alloc(NE * 2);
  _Float16* khB  = (_Float16*)alloc(NE * 2);
  _Float16* vTB  = (_Float16*)alloc(NE * 2);
  float*    qf32 = (float*)   alloc(NE * 4);
  float*    yatt = (float*)   alloc(NE * 4);
  _Float16* xT2  = (_Float16*)alloc(NE * 2);
  _Float16* wq16 = (_Float16*)alloc((size_t)C_ * C_ * 2);
  _Float16* wk16 = (_Float16*)alloc((size_t)C_ * C_ * 2);
  _Float16* wv16 = (_Float16*)alloc((size_t)C_ * C_ * 2);
  _Float16* wj16 = (_Float16*)alloc((size_t)C_ * C_ * 2);

  // 1) pack inputs + weights to f16
  pack_xT_kernel<<<2304, 256, 0, stream>>>(q, xTq);
  pack_xT_kernel<<<2304, 256, 0, stream>>>(k, xTk);
  pack_xT_kernel<<<2304, 256, 0, stream>>>(v, xTv);
  cvt_w16_kernel<<<64, 256, 0, stream>>>(wq,  wq16, C_ * C_);
  cvt_w16_kernel<<<64, 256, 0, stream>>>(wk,  wk16, C_ * C_);
  cvt_w16_kernel<<<64, 256, 0, stream>>>(wv,  wv16, C_ * C_);
  cvt_w16_kernel<<<64, 256, 0, stream>>>(wpj, wj16, C_ * C_);

  // 2) q/k/v projections (WMMA GEMM + BN) into attention-friendly layouts
  const int convBlocks = (BN_ * (C_ / 16) * (P_ / 16)) / 8;  // 1152
  conv1x1_bn_wmma<<<convBlocks, 256, 0, stream>>>(xTq, wq16, qg, qb2, qm, qv2,
                                                  qhB, nullptr, qf32);
  conv1x1_bn_wmma<<<convBlocks, 256, 0, stream>>>(xTk, wk16, kg, kb2, km, kv2,
                                                  khB, nullptr, nullptr);
  conv1x1_bn_wmma<<<convBlocks, 256, 0, stream>>>(xTv, wv16, vg, vb2, vm, vv2,
                                                  nullptr, vTB, nullptr);

  // 3) flash attention (WMMA QK^T + online softmax + WMMA PV)
  flash_attn_wmma<<<(BN_ * HEADS_ * (P_ / 16)) / 4, 128, 0, stream>>>(qhB, khB, vTB, yatt);

  // 4) positional depthwise conv + BN, add attention output, repack to f16
  pe_dwconv_add_pack<<<2304, 256, 0, stream>>>(qf32, pew, pg, pb2, pm, pv2, yatt, xT2);

  // 5) output projection (WMMA GEMM + BN) -> f32 NCHW
  conv1x1_bn_wmma<<<convBlocks, 256, 0, stream>>>(xT2, wj16, jg, jb2, jm, jv2,
                                                  nullptr, nullptr, (float*)d_out);
}
